// Net_49744311222864
// MI455X (gfx1250) — compile-verified, matrix operand order checked by reading, not run on hardware
//
#include <hip/hip_runtime.h>
#include <hip/hip_bf16.h>

typedef __attribute__((ext_vector_type(16))) _Float16 v16h;
typedef __attribute__((ext_vector_type(8)))  float    v8f;

#define NNODES 100000
#define NEDGES 1600000
#define NGRAPHS 1024
#define HDIM 64
#define H2 128
#define GEPS 1e-7f

// ---------------------------------------------------------------- zero
__global__ void zero_kernel(float* __restrict__ p, long long n) {
  long long i = (long long)blockIdx.x * blockDim.x + threadIdx.x;
  long long stride = (long long)gridDim.x * blockDim.x;
  for (; i < n; i += stride) p[i] = 0.0f;
}

// ---------------------------------------------------------------- node embed: h = x @ Wn + bn
__global__ void node_embed_kernel(const float* __restrict__ x,
                                  const float* __restrict__ W,
                                  const float* __restrict__ b,
                                  float* __restrict__ h) {
  __shared__ float Ws[16 * 64];
  __shared__ float bs[64];
  int tid = threadIdx.x;
  for (int i = tid; i < 16 * 64; i += 256) Ws[i] = W[i];
  if (tid < 64) bs[tid] = b[tid];
  __syncthreads();
  int gid = blockIdx.x * 256 + tid;
  int n = gid >> 6, f = gid & 63;
  if (n >= NNODES) return;
  const float* xr = x + (size_t)n * 16;
  float acc = bs[f];
  #pragma unroll
  for (int k = 0; k < 16; ++k) acc += xr[k] * Ws[k * 64 + f];
  h[(size_t)n * 64 + f] = acc;
}

// ---------------------------------------------------------------- edge pass 1: segment max
// msg = relu(h[src] + ea) + eps  (> 0, so int-pattern atomicMax is order-preserving; m init = +0.0)
__global__ void edge_max_kernel(const float* __restrict__ h,
                                const float* __restrict__ eattr,
                                const int* __restrict__ eidx,
                                const float* __restrict__ We,
                                const float* __restrict__ be,
                                float* __restrict__ m) {
  __shared__ float Ws[8 * 64];
  __shared__ float bs[64];
  int tid = threadIdx.x;
  for (int i = tid; i < 8 * 64; i += 256) Ws[i] = We[i];
  if (tid < 64) bs[tid] = be[tid];
  __syncthreads();
  int e = blockIdx.x * 16 + (tid >> 4);
  int q = tid & 15;
  if (e >= NEDGES) return;
  int s = eidx[e];
  int d = eidx[NEDGES + e];
  const float4* ep = (const float4*)(eattr + (size_t)e * 8);
  float4 e0 = ep[0], e1 = ep[1];
  float ev[8] = {e0.x, e0.y, e0.z, e0.w, e1.x, e1.y, e1.z, e1.w};
  float4 hv = ((const float4*)(h + (size_t)s * 64))[q];
  float hvv[4] = {hv.x, hv.y, hv.z, hv.w};
  int f0 = q * 4;
  int* md = (int*)(m + (size_t)d * 64 + f0);
  #pragma unroll
  for (int j = 0; j < 4; ++j) {
    int f = f0 + j;
    float ea = bs[f];
    #pragma unroll
    for (int k = 0; k < 8; ++k) ea += ev[k] * Ws[k * 64 + f];
    float msg = fmaxf(hvv[j] + ea, 0.0f) + GEPS;
    atomicMax(md + j, __float_as_int(msg));
  }
}

// ---------------------------------------------------------------- edge pass 2: denom + numerator
// e = exp(msg - m[dst]);  denom += e;  num += msg * e   (agg = num / max(denom,1e-16))
__global__ void edge_sum_kernel(const float* __restrict__ h,
                                const float* __restrict__ eattr,
                                const int* __restrict__ eidx,
                                const float* __restrict__ We,
                                const float* __restrict__ be,
                                const float* __restrict__ m,
                                float* __restrict__ den,
                                float* __restrict__ num) {
  __shared__ float Ws[8 * 64];
  __shared__ float bs[64];
  int tid = threadIdx.x;
  for (int i = tid; i < 8 * 64; i += 256) Ws[i] = We[i];
  if (tid < 64) bs[tid] = be[tid];
  __syncthreads();
  int e = blockIdx.x * 16 + (tid >> 4);
  int q = tid & 15;
  if (e >= NEDGES) return;
  int s = eidx[e];
  int d = eidx[NEDGES + e];
  const float4* ep = (const float4*)(eattr + (size_t)e * 8);
  float4 e0 = ep[0], e1 = ep[1];
  float ev[8] = {e0.x, e0.y, e0.z, e0.w, e1.x, e1.y, e1.z, e1.w};
  float4 hv = ((const float4*)(h + (size_t)s * 64))[q];
  float hvv[4] = {hv.x, hv.y, hv.z, hv.w};
  float4 mv = ((const float4*)(m + (size_t)d * 64))[q];
  float mvv[4] = {mv.x, mv.y, mv.z, mv.w};
  int f0 = q * 4;
  float* dd = den + (size_t)d * 64 + f0;
  float* nd = num + (size_t)d * 64 + f0;
  #pragma unroll
  for (int j = 0; j < 4; ++j) {
    int f = f0 + j;
    float ea = bs[f];
    #pragma unroll
    for (int k = 0; k < 8; ++k) ea += ev[k] * Ws[k * 64 + f];
    float msg = fmaxf(hvv[j] + ea, 0.0f) + GEPS;
    float ex = __expf(msg - mvv[j]);
    atomicAdd(dd + j, ex);
    atomicAdd(nd + j, msg * ex);
  }
}

// ---------------------------------------------------------------- node MLP (WMMA f16->f32)
// out = h + num/max(den,1e-16);  hid = relu(out@W1+b1);  h' = relu(hid@W2+b2)
// 256 threads = 8 waves, 32 nodes per block (two 16-row M tiles).
__global__ void __launch_bounds__(256)
node_mlp_kernel(const float* __restrict__ h_in,
                const float* __restrict__ numb,
                const float* __restrict__ denb,
                const float* __restrict__ W1, const float* __restrict__ b1,
                const float* __restrict__ W2, const float* __restrict__ b2,
                float* __restrict__ h_out) {
  __shared__ _Float16 W1t[H2 * HDIM];   // [n][k], k contiguous
  __shared__ _Float16 W2t[HDIM * H2];   // [n][k]
  __shared__ _Float16 Xa[32 * HDIM];    // A for GEMM1, row-major
  __shared__ _Float16 Hid[32 * H2];     // A for GEMM2, row-major
  __shared__ float b1s[H2];
  __shared__ float b2s[HDIM];
  int tid = threadIdx.x;
  for (int i = tid; i < HDIM * H2; i += 256) {       // W1 [64,128] -> W1t
    int k = i / H2, n = i % H2;
    W1t[n * HDIM + k] = (_Float16)W1[i];
  }
  for (int i = tid; i < H2 * HDIM; i += 256) {       // W2 [128,64] -> W2t
    int k = i / HDIM, n = i % HDIM;
    W2t[n * H2 + k] = (_Float16)W2[i];
  }
  if (tid < H2) b1s[tid] = b1[tid];
  if (tid < HDIM) b2s[tid] = b2[tid];
  int base = blockIdx.x * 32;
  for (int i = tid; i < 32 * HDIM; i += 256) {
    int n = base + (i >> 6);
    float v = 0.0f;
    if (n < NNODES) {
      size_t off = (size_t)n * 64 + (i & 63);
      v = h_in[off] + numb[off] / fmaxf(denb[off], 1e-16f);
    }
    Xa[i] = (_Float16)v;
  }
  __syncthreads();

  int wave = tid >> 5, lane = tid & 31;
  int half = lane >> 4, lm = lane & 15;

  // ---- GEMM1: [32,64] x [64,128] ; 16 (mt,nt) tiles over 8 waves, K = 2x32
  #pragma unroll
  for (int rep = 0; rep < 2; ++rep) {
    int p = wave * 2 + rep;
    int mt = p >> 3, nt = p & 7;
    v8f c = {};
    #pragma unroll
    for (int ks = 0; ks < 2; ++ks) {
      const _Float16* arow = &Xa[(mt * 16 + lm) * HDIM + ks * 32 + 8 * half];
      const _Float16* brow = &W1t[(nt * 16 + lm) * HDIM + ks * 32 + 8 * half];
      v16h a, b;
      #pragma unroll
      for (int v = 0; v < 8; ++v) {
        int kb = (v < 4) ? 2 * v : 2 * v + 8;   // ISA 16-bit A/B K layout
        a[2 * v]     = arow[kb];
        a[2 * v + 1] = arow[kb + 1];
        b[2 * v]     = brow[kb];
        b[2 * v + 1] = brow[kb + 1];
      }
      c = __builtin_amdgcn_wmma_f32_16x16x32_f16(false, a, false, b,
                                                 (short)0, c, false, false);
    }
    int gn = nt * 16 + lm;
    #pragma unroll
    for (int r = 0; r < 8; ++r) {
      int row = mt * 16 + r + 8 * half;
      float val = fmaxf(c[r] + b1s[gn], 0.0f);
      Hid[row * H2 + gn] = (_Float16)val;
    }
  }
  __syncthreads();

  // ---- GEMM2: [32,128] x [128,64] ; 8 tiles, one per wave, K = 4x32
  {
    int mt = wave >> 2, nt = wave & 3;
    v8f c = {};
    #pragma unroll
    for (int ks = 0; ks < 4; ++ks) {
      const _Float16* arow = &Hid[(mt * 16 + lm) * H2 + ks * 32 + 8 * half];
      const _Float16* brow = &W2t[(nt * 16 + lm) * H2 + ks * 32 + 8 * half];
      v16h a, b;
      #pragma unroll
      for (int v = 0; v < 8; ++v) {
        int kb = (v < 4) ? 2 * v : 2 * v + 8;
        a[2 * v]     = arow[kb];
        a[2 * v + 1] = arow[kb + 1];
        b[2 * v]     = brow[kb];
        b[2 * v + 1] = brow[kb + 1];
      }
      c = __builtin_amdgcn_wmma_f32_16x16x32_f16(false, a, false, b,
                                                 (short)0, c, false, false);
    }
    int gn = nt * 16 + lm;
    #pragma unroll
    for (int r = 0; r < 8; ++r) {
      int node = base + mt * 16 + r + 8 * half;
      if (node < NNODES) {
        float val = fmaxf(c[r] + b2s[gn], 0.0f);   // outer relu from reference
        h_out[(size_t)node * 64 + gn] = val;
      }
    }
  }
}

// ---------------------------------------------------------------- mean pool accumulate
__global__ void pool_kernel(const float* __restrict__ h,
                            const int* __restrict__ batch,
                            float* __restrict__ gsum,
                            float* __restrict__ gcnt) {
  int gid = blockIdx.x * 256 + threadIdx.x;
  int n = gid >> 6, f = gid & 63;
  if (n >= NNODES) return;
  int g = batch[n];
  atomicAdd(&gsum[g * 64 + f], h[(size_t)n * 64 + f]);
  if (f == 0) atomicAdd(&gcnt[g], 1.0f);
}

// ---------------------------------------------------------------- dense head, 1 thread / graph
__global__ void head_kernel(const float* __restrict__ gsum,
                            const float* __restrict__ gcnt,
                            const float* __restrict__ gattr,
                            const float* __restrict__ d1W, const float* __restrict__ d1b,
                            const float* __restrict__ d2W, const float* __restrict__ d2b,
                            const float* __restrict__ oW,  const float* __restrict__ ob,
                            float* __restrict__ out) {
  int g = blockIdx.x * 256 + threadIdx.x;
  if (g >= NGRAPHS) return;
  float in[74];
  float inv = 1.0f / fmaxf(gcnt[g], 1.0f);
  #pragma unroll
  for (int f = 0; f < 64; ++f) in[f] = gsum[g * 64 + f] * inv;
  #pragma unroll
  for (int j = 0; j < 10; ++j) in[64 + j] = gattr[g * 10 + j];
  float t1[32];
  for (int o = 0; o < 32; ++o) {
    float acc = d1b[o];
    for (int i = 0; i < 74; ++i) acc += in[i] * d1W[i * 32 + o];
    t1[o] = fmaxf(acc, 0.0f);
  }
  float t2[32];
  for (int o = 0; o < 32; ++o) {
    float acc = d2b[o];
    for (int i = 0; i < 32; ++i) acc += t1[i] * d2W[i * 32 + o];
    t2[o] = fmaxf(acc, 0.0f);
  }
  float z = ob[0];
  for (int i = 0; i < 32; ++i) z += t2[i] * oW[i];
  out[g] = 1.0f / (1.0f + __expf(-z));
}

// ----------------------------------------------------------------
extern "C" void kernel_launch(void* const* d_in, const int* in_sizes, int n_in,
                              void* d_out, int out_size, void* d_ws, size_t ws_size,
                              hipStream_t stream) {
  const float* x     = (const float*)d_in[0];
  const float* eattr = (const float*)d_in[1];
  const float* gattr = (const float*)d_in[2];
  const int*   eidx  = (const int*)d_in[3];
  const int*   batch = (const int*)d_in[4];
  const float* nodeW = (const float*)d_in[5];
  const float* nodeb = (const float*)d_in[6];
  const float* edgeW = (const float*)d_in[7];
  const float* edgeb = (const float*)d_in[8];
  const float* cW1[3] = {(const float*)d_in[9],  (const float*)d_in[13], (const float*)d_in[17]};
  const float* cb1[3] = {(const float*)d_in[10], (const float*)d_in[14], (const float*)d_in[18]};
  const float* cW2[3] = {(const float*)d_in[11], (const float*)d_in[15], (const float*)d_in[19]};
  const float* cb2[3] = {(const float*)d_in[12], (const float*)d_in[16], (const float*)d_in[20]};
  const float* d1W = (const float*)d_in[21]; const float* d1b = (const float*)d_in[22];
  const float* d2W = (const float*)d_in[23]; const float* d2b = (const float*)d_in[24];
  const float* oW  = (const float*)d_in[25]; const float* ob  = (const float*)d_in[26];

  // workspace: entire node-resident state ~128.3 MB (fits MI455X 192MB L2)
  char* ws = (char*)d_ws;
  size_t NH = (size_t)NNODES * 64;
  float* h_a  = (float*)ws; ws += NH * 4;
  float* h_b  = (float*)ws; ws += NH * 4;
  float* mbuf = (float*)ws; ws += NH * 4;   // m, den, num contiguous -> one zero pass
  float* den  = (float*)ws; ws += NH * 4;
  float* numb = (float*)ws; ws += NH * 4;
  float* gsum = (float*)ws; ws += (size_t)NGRAPHS * 64 * 4;
  float* gcnt = (float*)ws; ws += (size_t)NGRAPHS * 4;

  dim3 blk(256);
  node_embed_kernel<<<(NNODES * 64 + 255) / 256, blk, 0, stream>>>(x, nodeW, nodeb, h_a);

  float* hin = h_a;
  float* hout = h_b;
  for (int L = 0; L < 3; ++L) {
    zero_kernel<<<2048, blk, 0, stream>>>(mbuf, (long long)NH * 3);
    edge_max_kernel<<<(NEDGES + 15) / 16, blk, 0, stream>>>(hin, eattr, eidx, edgeW, edgeb, mbuf);
    edge_sum_kernel<<<(NEDGES + 15) / 16, blk, 0, stream>>>(hin, eattr, eidx, edgeW, edgeb,
                                                            mbuf, den, numb);
    node_mlp_kernel<<<(NNODES + 31) / 32, blk, 0, stream>>>(hin, numb, den,
                                                            cW1[L], cb1[L], cW2[L], cb2[L], hout);
    float* t = hin; hin = hout; hout = t;
  }
  // final node features now in hin
  zero_kernel<<<64, blk, 0, stream>>>(gsum, (long long)NGRAPHS * 64 + NGRAPHS);
  pool_kernel<<<(NNODES * 64 + 255) / 256, blk, 0, stream>>>(hin, batch, gsum, gcnt);
  head_kernel<<<(NGRAPHS + 255) / 256, blk, 0, stream>>>(gsum, gcnt, gattr,
                                                         d1W, d1b, d2W, d2b, oW, ob,
                                                         (float*)d_out);
}
